// SpatDivisiveNorm_56916906606729
// MI455X (gfx1250) — matrix operand order repaired and checked by reading, not compiled, not run on hardware
//
#include <hip/hip_runtime.h>
#include <hip/hip_bf16.h>

// ---------------------------------------------------------------------------
// SpatDivisiveNorm on MI455X (gfx1250).
// conv3d(x^2, f(|dc|)*g(r)) factored into channel-conv (65 taps, circular over
// 65 with one zero slot) followed by spatial-conv (129x129, zero-padded),
// mapped to v_wmma_f32_16x16x32_bf16 via per-dh Toeplitz operands.
// ---------------------------------------------------------------------------

typedef __attribute__((ext_vector_type(16))) __bf16 v16bf;
typedef __attribute__((ext_vector_type(8)))  float  v8f;

union Frag16 { v16bf v; uint4 q[2]; };

#define BT_ELEMS  (129 * 5 * 32 * 16)        // 330240 bf16: [dh][kb][lane][16]
#define WS_BT_OFF 512                        // bytes
#define WS_S_OFF  (512 + 660480)             // bytes (BT_ELEMS*2 = 660480)
#define PATCH_COLS 176                       // bf16 elems per patch row (16B mult)
#define PATCH_ROWS 160

// ---------------------------------------------------------------------------
// Kernel W: build channel filter table (norm folded in) and WMMA-ready
// Toeplitz B operands for every dh. B[k,n] = g(dh-64, (k-n)-64), k-n in [0,128].
// Lane layout for 16-bit B (32x16): n = lane&15, K = (lane>>4)*8 + i (i<8), 16 + (lane>>4)*8 + i-8 (i>=8).
// ---------------------------------------------------------------------------
__global__ __launch_bounds__(256) void sdn_weights_kernel(
    const float* __restrict__ lamb, const float* __restrict__ xy_lamb,
    float* __restrict__ ftab, __hip_bfloat16* __restrict__ Btab) {
  int gid = blockIdx.x * 256 + threadIdx.x;
  float la  = lamb[0];
  float xla = xy_lamb[0];
  if (gid < 65) {
    float inv = 1.0f / ((la + 1e-6f) * (xla + 1e-6f) * (xla + 1e-6f));
    ftab[gid] = __expf(-fabsf((float)(gid - 32)) / la) * inv;
  }
  if (gid < BT_ELEMS) {
    int dh   = gid / 2560;          // 5*32*16
    int r    = gid - dh * 2560;
    int kb   = r >> 9;              // / 512
    int lane = (r >> 4) & 31;
    int i    = r & 15;
    int n    = lane & 15;
    int khi  = lane >> 4;
    int kk   = kb * 32 + (i < 8 ? khi * 8 + i : 16 + khi * 8 + (i - 8));
    int j    = kk - n;              // dw index, valid [0,128]
    float val = 0.0f;
    if (j >= 0 && j <= 128) {
      float dhc = (float)(dh - 64);
      float dwc = (float)(j - 64);
      val = __expf(-sqrtf(dhc * dhc + dwc * dwc) / xla);
    }
    Btab[gid] = __float2bfloat16(val);
  }
}

// ---------------------------------------------------------------------------
// Kernel S: s[b,c,h,w] = sum_{dc=-32..32} f[dc+32] * x^2[b,(c+dc) mod 65,h,w]
// with channel 64 == zero pad (reproduces the reference's circular wrap).
// One block per (b,h): stage all 64 channels of one row in LDS.
// ---------------------------------------------------------------------------
__global__ __launch_bounds__(256) void sdn_chan_kernel(
    const float* __restrict__ x, const float* __restrict__ ftab,
    __hip_bfloat16* __restrict__ sbuf) {
  __shared__ float X2[64 * 128];
  __shared__ float fs[65];
  int h = blockIdx.x, b = blockIdx.y;
  int tid = threadIdx.x;
  if (tid < 65) fs[tid] = ftab[tid];
  for (int idx = tid; idx < 64 * 128; idx += 256) {
    int c = idx >> 7, w = idx & 127;
    float v = x[(((b << 6) + c) * 128 + h) * 128 + w];
    X2[idx] = v * v;
  }
  __syncthreads();
  for (int idx = tid; idx < 64 * 128; idx += 256) {
    int c = idx >> 7, w = idx & 127;
    float sum = 0.0f;
    for (int t = 0; t < 65; ++t) {
      int j = c - 32 + t;
      j = (j < 0) ? j + 65 : (j >= 65 ? j - 65 : j);
      if (j < 64) sum += fs[t] * X2[(j << 7) | w];
    }
    sbuf[(((b << 6) + c) * 128 + h) * 128 + w] = __float2bfloat16(sum);
  }
}

// ---------------------------------------------------------------------------
// Kernel C: spatial conv via WMMA + finalize.
// Block = 128 threads = 4 waves = 2x2 tiles of 16x16 outputs (32x32 region).
// LDS patch 160x176 bf16 (zeroed borders). Per dh (129): A = 16x160 slab from
// LDS (two ds_load_b128 per lane per kb), B = Toeplitz frag from L2-resident
// Btab, 5x wmma_f32_16x16x32_bf16 accumulate.
// ---------------------------------------------------------------------------
__global__ __launch_bounds__(128) void sdn_wmma_kernel(
    const __hip_bfloat16* __restrict__ sbuf,
    const __hip_bfloat16* __restrict__ Btab,
    const float* __restrict__ x,
    const float* __restrict__ alpha, const float* __restrict__ beta,
    const float* __restrict__ kp, float* __restrict__ out) {
  __shared__ __hip_bfloat16 patch[PATCH_ROWS * PATCH_COLS];  // 56320 B

  int tid = threadIdx.x;
  int img = blockIdx.y;
  int blockH = (blockIdx.x >> 2) * 32;
  int blockW = (blockIdx.x & 3) * 32;

  // zero LDS (borders stay zero)
  uint* pz = (uint*)patch;
  for (int i = tid; i < PATCH_ROWS * PATCH_COLS / 2; i += 128) pz[i] = 0u;
  __syncthreads();

  // fill interior: global rows [blockH-64, blockH+96) ∩ [0,128),
  // cols always the valid slice of [blockW-64, blockW+112); widths are
  // multiples of 8 and 16B-aligned both sides -> uint4 copies.
  int gstart = blockW - 64 < 0 ? 0 : blockW - 64;
  int gend   = blockW + 112 > 128 ? 128 : blockW + 112;
  int nq     = (gend - gstart) >> 3;
  int dstart = gstart - blockW + 64;
  for (int i = tid; i < PATCH_ROWS * 16; i += 128) {
    int r = i >> 4, qi = i & 15;
    int gr = blockH - 64 + r;
    if (gr >= 0 && gr < 128 && qi < nq) {
      const uint4* src = (const uint4*)(sbuf + (img * 128 + gr) * 128 + gstart);
      *(uint4*)(&patch[r * PATCH_COLS + dstart + qi * 8]) = src[qi];
    }
  }
  __syncthreads();

  int wave = tid >> 5, lane = tid & 31;
  int th = wave >> 1, tw = wave & 1;
  int m  = lane & 15, hi = lane >> 4;
  // 16-bit A (16x32) layout: lane m holds row m; elems 0..7 -> K=hi*8+0..7,
  // elems 8..15 -> K=16+hi*8+0..7  => two 16B chunks at col +0 and +16.
  int rowbase = th * 16 + m;
  int cb      = tw * 16 + (hi << 3);

  v8f acc = {0.f, 0.f, 0.f, 0.f, 0.f, 0.f, 0.f, 0.f};

  for (int dh = 0; dh < 129; ++dh) {
    const __hip_bfloat16* prow = &patch[(rowbase + dh) * PATCH_COLS + cb];
    if (dh < 128)  // keep next dh's Toeplitz operand warm
      __builtin_prefetch(Btab + (((dh + 1) * 5) * 32 + lane) * 16, 0, 1);
#pragma unroll
    for (int kb = 0; kb < 5; ++kb) {
      Frag16 a;
      a.q[0] = *(const uint4*)(prow + kb * 32);
      a.q[1] = *(const uint4*)(prow + kb * 32 + 16);
      Frag16 bfr;
      const uint4* bp = (const uint4*)(Btab + ((dh * 5 + kb) * 32 + lane) * 16);
      bfr.q[0] = bp[0];
      bfr.q[1] = bp[1];
      acc = __builtin_amdgcn_wmma_f32_16x16x32_bf16(
          false, a.v, false, bfr.v, (short)0, acc, false, false);
    }
  }

  // finalize: D layout lanes 0-15 -> (M=r, N=lane); lanes 16-31 -> (M=8+r).
  float al = alpha[0], be = beta[0], kk = kp[0];
  int h0 = blockH + th * 16, w0 = blockW + tw * 16;
  int wcol = w0 + m;
#pragma unroll
  for (int r = 0; r < 8; ++r) {
    int hrow = h0 + (hi << 3) + r;
    int gidx = (img * 128 + hrow) * 128 + wcol;
    float xv = x[gidx];
    float v  = acc[r] * al + kk;             // conv already normalized (ftab)
    float den = __expf(be * __logf(v));      // v >= k > 0
    out[gidx] = xv * xv / (den + 1e-6f);
  }
}

// ---------------------------------------------------------------------------
extern "C" void kernel_launch(void* const* d_in, const int* in_sizes, int n_in,
                              void* d_out, int out_size, void* d_ws, size_t ws_size,
                              hipStream_t stream) {
  const float* x       = (const float*)d_in[0];
  const float* lamb    = (const float*)d_in[1];
  const float* xy_lamb = (const float*)d_in[2];
  const float* alpha   = (const float*)d_in[3];
  const float* beta    = (const float*)d_in[4];
  const float* kp      = (const float*)d_in[5];
  float* out = (float*)d_out;

  char* ws = (char*)d_ws;
  float*           ftab = (float*)ws;
  __hip_bfloat16*  Btab = (__hip_bfloat16*)(ws + WS_BT_OFF);
  __hip_bfloat16*  sbuf = (__hip_bfloat16*)(ws + WS_S_OFF);  // 512*128*128 bf16

  sdn_weights_kernel<<<dim3((BT_ELEMS + 255) / 256), dim3(256), 0, stream>>>(
      lamb, xy_lamb, ftab, Btab);
  sdn_chan_kernel<<<dim3(128, 8), dim3(256), 0, stream>>>(x, ftab, sbuf);
  sdn_wmma_kernel<<<dim3(16, 512), dim3(128), 0, stream>>>(
      sbuf, Btab, x, alpha, beta, kp, out);
}
